// DLDMD_16449724745509
// MI455X (gfx1250) — compile-verified, matrix operand order checked by reading, not run on hardware
//
#include <hip/hip_runtime.h>

typedef __attribute__((ext_vector_type(16))) _Float16 v16h;
typedef __attribute__((ext_vector_type(8)))  float    v8f;

// ---- problem geometry ----
#define NBATCH 512
#define TLEN   256
#define NROWS  131072            // B*T
// ---- d_out float offsets (tuple flattened, complex64 = 2 floats) ----
#define Y_OFF     0u
#define XAE_OFF   4194304u
#define XADV_OFF  4587520u
#define YADV_OFF  4980736u
#define EVALS_OFF 9175040u
#define EVECS_OFF 9207808u
#define PHI_OFF   10256384u
// ---- workspace (in _Float16 units) ----
#define WS_ENC_H 0u              // 4 * 65536
#define WS_DEC_H 262144u         // 4 * 65536
#define WS_ENC_O 524288u         // 8 kt * 2 nt * 512
#define WS_DEC_I 532480u         // 1 kt * 16 nt * 512
// ---- dynamic LDS layout for MLP kernels ----
#define LDSA_BYTES 33792u        // 4 waves * 16 rows * 264 halves * 2B
#define LDSB_BYTES 131072u       // one hidden layer of packed B fragments
#define MLP_LDS_BYTES (LDSA_BYTES + LDSB_BYTES)   // 164,864 B (<320KB/WGP)

// ================= complex helpers =================
__device__ __forceinline__ float2 cmul(float2 a, float2 b){ return make_float2(a.x*b.x-a.y*b.y, a.x*b.y+a.y*b.x); }
__device__ __forceinline__ float2 cadd(float2 a, float2 b){ return make_float2(a.x+b.x, a.y+b.y); }
__device__ __forceinline__ float2 csub(float2 a, float2 b){ return make_float2(a.x-b.x, a.y-b.y); }
__device__ __forceinline__ float2 cconj(float2 a){ return make_float2(a.x, -a.y); }
__device__ __forceinline__ float  cabs2(float2 a){ return a.x*a.x + a.y*a.y; }
__device__ __forceinline__ float2 cscale(float2 a, float s){ return make_float2(a.x*s, a.y*s); }
__device__ __forceinline__ float2 cdiv(float2 a, float2 b){
  float d = cabs2(b) + 1e-30f;
  return make_float2((a.x*b.x + a.y*b.y)/d, (a.y*b.x - a.x*b.y)/d);
}
__device__ __forceinline__ float2 csqrtc(float2 z){
  float r  = sqrtf(sqrtf(cabs2(z)));
  float th = 0.5f*atan2f(z.y, z.x);
  return make_float2(r*cosf(th), r*sinf(th));
}
__device__ __forceinline__ v8f splat8(float x){
  v8f r;
  #pragma unroll
  for (int i=0;i<8;i++) r[i]=x;
  return r;
}

// ====== CDNA5 async global->LDS staging (ASYNCcnt path, b128 granules) ======
// Copies 131072 bytes (one packed hidden layer) into LDS, 16B per lane-op.
__device__ __forceinline__ void stage_layer_async(const _Float16* __restrict__ g,
                                                  unsigned lds_base_bytes, int tid)
{
  #pragma unroll 8
  for (int it = tid; it < 8192; it += 128){
    unsigned    laddr = lds_base_bytes + (unsigned)it*16u;
    const char* gp    = (const char*)g + (size_t)it*16u;
    asm volatile("global_load_async_to_lds_b128 %0, %1, off"
                 :: "v"(laddr), "v"(gp) : "memory");
  }
  asm volatile("s_wait_asynccnt 0x0" ::: "memory");
}

// ================= weight packing into WMMA B-fragment layout =================
// W is (K,N) row-major f32. One block (32 threads) packs one 32x16 fragment.
// Fragment element e of lane L holds W[kt*32 + ((e<8)?0:16) + (L/16)*8 + (e&7)][nt*16 + (L&15)].
__global__ void pack_w_f16(const float* __restrict__ W, int K, int N,
                           _Float16* __restrict__ dst)
{
  (void)K;
  const int kt = blockIdx.x, nt = blockIdx.y, NT = gridDim.y;
  const int lane = threadIdx.x;
  _Float16* out = dst + ((size_t)(kt*NT + nt))*512 + (size_t)lane*16;
  const int n = nt*16 + (lane & 15);
  const int kb = kt*32 + ((lane >> 4) << 3);
  #pragma unroll
  for (int e=0; e<16; e++){
    int k = kb + ((e < 8) ? 0 : 16) + (e & 7);
    out[e] = (_Float16)W[(size_t)k*N + n];
  }
}

// ================= shared hidden-layer (256->256 tanh) via WMMA =================
// One wave owns a 16x256 activation tile in LDS (row stride 264 halves).
// B fragments come from LDS (staged by stage_layer_async), double-buffered in
// registers so the ds_load for K-step kt+1 overlaps the WMMA of step kt.
__device__ __forceinline__ void hidden_layer(_Float16 (*Atile)[264],
                                             const _Float16* ldsB,
                                             const float* __restrict__ bl,
                                             int lane)
{
  const int lr = lane & 15, lh = lane >> 4;
  v16h afrag[8];
  #pragma unroll
  for (int kt=0; kt<8; kt++){
    union { v16h v; uint4 q[2]; } u;
    u.q[0] = *(const uint4*)&Atile[lr][kt*32 + lh*8];
    u.q[1] = *(const uint4*)&Atile[lr][kt*32 + 16 + lh*8];
    afrag[kt] = u.v;
  }
  float bias[16];
  #pragma unroll
  for (int nt=0; nt<16; nt++) bias[nt] = bl[nt*16 + lr];

  union BU { v16h v; uint4 q[2]; };
  for (int nt=0; nt<16; nt++){
    v8f acc = splat8(bias[nt]);
    const _Float16* base = ldsB + (size_t)nt*512 + (size_t)lane*16;   // kt stride = 16*512 halves
    BU bufs[2];
    bufs[0].q[0] = ((const uint4*)base)[0];
    bufs[0].q[1] = ((const uint4*)base)[1];
    #pragma unroll
    for (int kt=0; kt<8; kt++){
      if (kt < 7){
        const uint4* np = (const uint4*)(base + (size_t)(kt+1)*16*512);
        bufs[(kt+1)&1].q[0] = np[0];
        bufs[(kt+1)&1].q[1] = np[1];
      }
      acc = __builtin_amdgcn_wmma_f32_16x16x32_f16(false, afrag[kt], false, bufs[kt&1].v,
                                                   (short)0, acc, false, false);
    }
    #pragma unroll
    for (int r=0; r<8; r++)
      Atile[r + lh*8][nt*16 + lr] = (_Float16)tanhf(acc[r]);
  }
}

// ================= encoder: x(131072x3) -> y(131072x32) =================
__global__ __launch_bounds__(128) void enc_kernel(
    const float* __restrict__ x,  const float* __restrict__ Wi, const float* __restrict__ bi,
    const _Float16* __restrict__ pH, const float* __restrict__ bh,
    const _Float16* __restrict__ pO, const float* __restrict__ bo,
    float* __restrict__ y)
{
  extern __shared__ char dynsmem[];
  _Float16 (*AtileAll)[16][264] = (_Float16(*)[16][264])dynsmem;
  _Float16* ldsB = (_Float16*)(dynsmem + LDSA_BYTES);
  const unsigned ldsB_base = (unsigned)(uintptr_t)ldsB;

  const int w = threadIdx.x >> 5, lane = threadIdx.x & 31;
  const int lr = lane & 15, lh = lane >> 4;
  const int rowBase = blockIdx.x*64 + w*16;
  _Float16 (*Atile)[264] = AtileAll[w];

  { // input layer 3->256, lane owns 8 columns
    float w0[8], w1[8], w2[8], bb[8];
    #pragma unroll
    for (int c8=0;c8<8;c8++){
      int c = lane + c8*32;
      w0[c8]=Wi[c]; w1[c8]=Wi[256+c]; w2[c8]=Wi[512+c]; bb[c8]=bi[c];
    }
    for (int r=0;r<16;r++){
      const float* xr = x + (size_t)(rowBase+r)*3;
      float x0=xr[0], x1=xr[1], x2=xr[2];
      #pragma unroll
      for (int c8=0;c8<8;c8++)
        Atile[r][lane + c8*32] = (_Float16)tanhf(x0*w0[c8]+x1*w1[c8]+x2*w2[c8]+bb[c8]);
    }
  }
  for (int layer=0; layer<4; layer++){
    __syncthreads();                                   // prior reads of ldsB done
    stage_layer_async(pH + (size_t)layer*65536, ldsB_base, threadIdx.x);
    __syncthreads();                                   // staged data visible to all waves
    hidden_layer(Atile, ldsB, bh + layer*256, lane);
  }
  __syncthreads();
  { // output layer 256->32 (2 n-tiles), no activation
    v16h afrag[8];
    #pragma unroll
    for (int kt=0;kt<8;kt++){
      union { v16h v; uint4 q[2]; } u;
      u.q[0] = *(const uint4*)&Atile[lr][kt*32 + lh*8];
      u.q[1] = *(const uint4*)&Atile[lr][kt*32 + 16 + lh*8];
      afrag[kt]=u.v;
    }
    float bias2[2];
    bias2[0]=bo[lr]; bias2[1]=bo[16+lr];
    for (int nt=0; nt<2; nt++){
      v8f acc = splat8(bias2[nt]);
      #pragma unroll
      for (int kt=0;kt<8;kt++){
        union { v16h v; uint4 q[2]; } bu;
        const uint4* bp = (const uint4*)(pO + ((size_t)(kt*2 + nt))*512 + (size_t)lane*16);
        bu.q[0]=bp[0]; bu.q[1]=bp[1];
        acc = __builtin_amdgcn_wmma_f32_16x16x32_f16(false, afrag[kt], false, bu.v,
                                                     (short)0, acc, false, false);
      }
      #pragma unroll
      for (int r=0;r<8;r++)
        y[(size_t)(rowBase + r + lh*8)*32 + nt*16 + lr] = acc[r];
    }
  }
}

// ================= decoder: in(131072x32) -> out(131072x3) =================
__global__ __launch_bounds__(128) void dec_kernel(
    const float* __restrict__ in,
    const _Float16* __restrict__ pI, const float* __restrict__ bin,
    const _Float16* __restrict__ pH, const float* __restrict__ bh,
    const float* __restrict__ Wout, const float* __restrict__ bout,
    float* __restrict__ out)
{
  extern __shared__ char dynsmem[];
  _Float16 (*AtileAll)[16][264] = (_Float16(*)[16][264])dynsmem;
  _Float16* ldsB = (_Float16*)(dynsmem + LDSA_BYTES);
  const unsigned ldsB_base = (unsigned)(uintptr_t)ldsB;

  const int w = threadIdx.x >> 5, lane = threadIdx.x & 31;
  const int lr = lane & 15, lh = lane >> 4;
  const int rowBase = blockIdx.x*64 + w*16;
  _Float16 (*Atile)[264] = AtileAll[w];

  { // input layer 32->256 : single K-tile WMMA per n-tile
    union { v16h v; } u;
    const float* ip = in + (size_t)(rowBase + lr)*32;
    #pragma unroll
    for (int e=0;e<8;e++) u.v[e]   = (_Float16)ip[lh*8 + e];
    #pragma unroll
    for (int e=0;e<8;e++) u.v[8+e] = (_Float16)ip[16 + lh*8 + e];
    float biasi[16];
    #pragma unroll
    for (int nt=0; nt<16; nt++) biasi[nt] = bin[nt*16 + lr];
    for (int nt=0; nt<16; nt++){
      v8f acc = splat8(biasi[nt]);
      union { v16h v; uint4 q[2]; } bu;
      const uint4* bp = (const uint4*)(pI + (size_t)nt*512 + (size_t)lane*16);
      bu.q[0]=bp[0]; bu.q[1]=bp[1];
      acc = __builtin_amdgcn_wmma_f32_16x16x32_f16(false, u.v, false, bu.v,
                                                   (short)0, acc, false, false);
      #pragma unroll
      for (int r=0;r<8;r++)
        Atile[r + lh*8][nt*16 + lr] = (_Float16)tanhf(acc[r]);
    }
  }
  for (int layer=0; layer<4; layer++){
    __syncthreads();
    stage_layer_async(pH + (size_t)layer*65536, ldsB_base, threadIdx.x);
    __syncthreads();
    hidden_layer(Atile, ldsB, bh + layer*256, lane);
  }
  __syncthreads();
  { // output layer 256->3 : lane lh halves of K, combine with shfl
    float s0=0.f, s1=0.f, s2=0.f;
    for (int k = lh*128; k < lh*128 + 128; k++){
      float hv = (float)Atile[lr][k];
      s0 += hv*Wout[k*3+0]; s1 += hv*Wout[k*3+1]; s2 += hv*Wout[k*3+2];
    }
    s0 += __shfl_xor(s0, 16, 32);
    s1 += __shfl_xor(s1, 16, 32);
    s2 += __shfl_xor(s2, 16, 32);
    if (lh == 0){
      float* o = out + (size_t)(rowBase + lr)*3;
      o[0]=s0+bout[0]; o[1]=s1+bout[1]; o[2]=s2+bout[2];
    }
  }
}

// ================= DMD: one wave (32 lanes) per batch element =================
__global__ __launch_bounds__(32) void dmd_kernel(
    const float* __restrict__ yall,
    float* __restrict__ evals_out, float* __restrict__ evecs_out,
    float* __restrict__ phi_out,   float* __restrict__ yadv_out)
{
  __shared__ float  S1[32][33], S2[32][33], S3[32][33], S4[32][33];
  __shared__ float2 C1[32][33], C2[32][33], RHS[32][33];
  __shared__ float2 evalS[32], y0S[32], cS[32], rv[32], xv[32], gs[32];
  __shared__ float  vv[32], gc[32], eigv[32];
  __shared__ int    piv[32];

  const int lane = threadIdx.x;
  const int b = blockIdx.x;
  const float* Yg = yall + (size_t)b*8192;   // Y[t][l]; Ym[i][t]=Yg[t*32+i], t<255

  // ---- G = Ym Ym^T, P = Yp Ym^T (lane = column j) ----
  for (int i=0;i<32;i++){
    float g=0.f, p=0.f;
    for (int t=0;t<255;t++){
      float yj  = Yg[t*32 + lane];
      float yi  = Yg[t*32 + i];
      float yi1 = Yg[(t+1)*32 + i];
      g += yi*yj;  p += yi1*yj;
    }
    S1[i][lane]=g; S3[i][lane]=p;
  }
  for (int i=0;i<32;i++) S2[i][lane] = (i==lane) ? 1.f : 0.f;
  __syncthreads();

  // ---- cyclic Jacobi on symmetric G ----
  for (int sw=0; sw<10; sw++)
    for (int p=0;p<31;p++)
      for (int q=p+1;q<32;q++){
        float apq=S1[p][q], app=S1[p][p], aqq=S1[q][q];
        if (fabsf(apq) > 1e-12f*(fabsf(app)+fabsf(aqq)) + 1e-30f){
          float tau=(aqq-app)/(2.f*apq);
          float t=(tau>=0.f?1.f:-1.f)/(fabsf(tau)+sqrtf(1.f+tau*tau));
          float c=rsqrtf(1.f+t*t), s=t*c;
          float Gp=S1[p][lane], Gq=S1[q][lane];
          S1[p][lane]=c*Gp-s*Gq;  S1[q][lane]=s*Gp+c*Gq;
          __syncthreads();
          float Cp=S1[lane][p], Cq=S1[lane][q];
          S1[lane][p]=c*Cp-s*Cq;  S1[lane][q]=s*Cp+c*Cq;
          float Up=S2[lane][p], Uq=S2[lane][q];
          S2[lane][p]=c*Up-s*Uq;  S2[lane][q]=s*Up+c*Uq;
          __syncthreads();
        }
      }
  eigv[lane]=S1[lane][lane];
  __syncthreads();
  float inv2 = eigv[lane] > 1e-18f ? 1.f/eigv[lane] : 0.f;

  // ---- A = P U diag(1/sig^2) U^T ----
  for (int i=0;i<32;i++){
    float a=0.f; for (int k=0;k<32;k++) a += S3[i][k]*S2[k][lane];
    S1[i][lane]=a*inv2;
  }
  __syncthreads();
  for (int i=0;i<32;i++){
    float a=0.f; for (int k=0;k<32;k++) a += S1[i][k]*S2[lane][k];
    S4[i][lane]=a;
  }
  __syncthreads();

  // ---- Hessenberg reduction (Householder) of copy A -> S3 ----
  for (int i=0;i<32;i++) S3[i][lane]=S4[i][lane];
  __syncthreads();
  for (int k=0;k<30;k++){
    float x0=S3[k+1][k], sg=0.f;
    for (int i=k+2;i<32;i++){ float xi=S3[i][k]; sg += xi*xi; }
    if (sg > 1e-30f){
      float mu = sqrtf(x0*x0 + sg);
      float v0 = (x0 <= 0.f) ? (x0 - mu) : (x0 + mu);
      float beta = 2.f*v0*v0/(sg + v0*v0);
      if (lane == k+1)      vv[lane]=1.f;
      else if (lane > k+1)  vv[lane]=S3[lane][k]/v0;
      else                  vv[lane]=0.f;
      __syncthreads();
      float wsum=0.f;
      for (int i=k+1;i<32;i++) wsum += vv[i]*S3[i][lane];
      for (int i=k+1;i<32;i++) S3[i][lane] -= beta*vv[i]*wsum;
      __syncthreads();
      float w2=0.f;
      for (int j=k+1;j<32;j++) w2 += S3[lane][j]*vv[j];
      for (int j=k+1;j<32;j++) S3[lane][j] -= beta*w2*vv[j];
      __syncthreads();
    }
  }
  // complex Hessenberg
  for (int i=0;i<32;i++) C1[i][lane]=make_float2(S3[i][lane], 0.f);
  __syncthreads();

  // ---- shifted complex QR with deflation ----
  int m=31;
  for (int guard=0; guard<2000; guard++){
    if (m <= 0) break;
    float2 off=C1[m][m-1];
    float sub=fabsf(off.x)+fabsf(off.y);
    float dg = fabsf(C1[m-1][m-1].x)+fabsf(C1[m-1][m-1].y)
             + fabsf(C1[m][m].x)+fabsf(C1[m][m].y);
    if (sub <= 1e-7f*dg + 1e-30f){
      if (lane==0) evalS[m]=C1[m][m];
      m--; continue;
    }
    float2 a=C1[m-1][m-1], b2=C1[m-1][m], c2=C1[m][m-1], d=C1[m][m];
    float2 hf=cscale(cadd(a,d),0.5f);
    float2 disc=csqrtc(csub(cmul(hf,hf), csub(cmul(a,d),cmul(b2,c2))));
    float2 l1=cadd(hf,disc), l2=csub(hf,disc);
    float2 mu = (cabs2(csub(l1,d)) < cabs2(csub(l2,d))) ? l1 : l2;
    if (lane <= m) C1[lane][lane]=csub(C1[lane][lane], mu);
    __syncthreads();
    for (int k=0;k<m;k++){           // QR row sweep
      float2 ha=C1[k][k], hb=C1[k+1][k];
      float na=sqrtf(cabs2(ha)), nb=sqrtf(cabs2(hb));
      float r=sqrtf(na*na+nb*nb);
      float c; float2 s;
      if (r < 1e-30f){ c=1.f; s=make_float2(0.f,0.f); }
      else if (na < 1e-30f){ c=0.f; s=cscale(cconj(hb),1.f/nb); }
      else { c=na/r; s=cscale(cmul(cscale(ha,1.f/na),cconj(hb)),1.f/r); }
      if (lane==0){ gc[k]=c; gs[k]=s; }
      float2 hk=C1[k][lane], hk1=C1[k+1][lane];
      C1[k][lane]   = cadd(cscale(hk,c),  cmul(s,hk1));
      C1[k+1][lane] = csub(cscale(hk1,c), cmul(cconj(s),hk));
      __syncthreads();
    }
    for (int k=0;k<m;k++){           // RQ column sweep (per-lane row)
      float c=gc[k]; float2 s=gs[k];
      float2 hk=C1[lane][k], hk1=C1[lane][k+1];
      C1[lane][k]   = cadd(cscale(hk,c),  cmul(cconj(s),hk1));
      C1[lane][k+1] = csub(cscale(hk1,c), cmul(s,hk));
    }
    __syncthreads();
    if (lane <= m) C1[lane][lane]=cadd(C1[lane][lane], mu);
    __syncthreads();
  }
  if (lane <= m) evalS[lane]=C1[lane][lane];
  __syncthreads();

  // ---- eigenvectors via inverse iteration (complex GE, rhs = ones) ----
  for (int j=0;j<32;j++){
    float2 lam=evalS[j];
    for (int i=0;i<32;i++){
      float2 v=make_float2(S4[i][lane],0.f);
      if (lane==i) v=csub(v,lam);
      C1[i][lane]=v;
    }
    rv[lane]=make_float2(1.f,0.f);
    __syncthreads();
    for (int k=0;k<32;k++){
      int pi=k; float best=cabs2(C1[k][k]);
      for (int i=k+1;i<32;i++){ float t=cabs2(C1[i][k]); if (t>best){best=t;pi=i;} }
      if (pi!=k){
        float2 t=C1[k][lane]; C1[k][lane]=C1[pi][lane]; C1[pi][lane]=t;
        if (lane==0){ float2 tt=rv[k]; rv[k]=rv[pi]; rv[pi]=tt; }
      }
      __syncthreads();
      if (lane==0 && cabs2(C1[k][k]) < 1e-28f) C1[k][k]=make_float2(1e-14f,0.f);
      __syncthreads();
      float2 pv=C1[k][k];
      for (int i=k+1;i<32;i++){
        float2 f=cdiv(C1[i][k], pv);
        if (lane>k) C1[i][lane]=csub(C1[i][lane], cmul(f, C1[k][lane]));
        if (lane==0) rv[i]=csub(rv[i], cmul(f, rv[k]));
      }
      __syncthreads();
    }
    for (int k=31;k>=0;k--){
      float2 sum=rv[k];
      for (int jj=k+1;jj<32;jj++) sum=csub(sum, cmul(C1[k][jj], xv[jj]));
      sum=cdiv(sum, C1[k][k]);
      if (lane==0) xv[k]=sum;
      __syncthreads();
    }
    float n2=0.f; for (int i=0;i<32;i++) n2 += cabs2(xv[i]);
    C2[lane][j]=cscale(xv[lane], rsqrtf(n2 + 1e-30f));
    __syncthreads();
  }

  // ---- LU of evecs (C2 -> C1) with pivots ----
  for (int i=0;i<32;i++) C1[i][lane]=C2[i][lane];
  __syncthreads();
  for (int k=0;k<32;k++){
    int pi=k; float best=cabs2(C1[k][k]);
    for (int i=k+1;i<32;i++){ float t=cabs2(C1[i][k]); if (t>best){best=t;pi=i;} }
    if (lane==0) piv[k]=pi;
    if (pi!=k){ float2 t=C1[k][lane]; C1[k][lane]=C1[pi][lane]; C1[pi][lane]=t; }
    __syncthreads();
    if (lane==0 && cabs2(C1[k][k]) < 1e-28f) C1[k][k]=make_float2(1e-14f,0.f);
    __syncthreads();
    float2 pv=C1[k][k];
    for (int i=k+1;i<32;i++){
      float2 f=cdiv(C1[i][k], pv);
      if (lane>k)  C1[i][lane]=csub(C1[i][lane], cmul(f, C1[k][lane]));
      if (lane==k) C1[i][k]=f;                   // store multiplier
    }
    __syncthreads();
  }

  // ---- phi = solve(evecs, Ym): batched, one RHS column per lane ----
  float2* po = (float2*)phi_out;
  for (int blk=0; blk<8; blk++){
    int col = blk*32 + lane;
    for (int i=0;i<32;i++)
      RHS[i][lane] = make_float2((col < 255) ? Yg[col*32 + i] : 0.f, 0.f);
    __syncthreads();
    for (int k=0;k<32;k++){
      int pk=piv[k];
      if (pk!=k){ float2 t=RHS[k][lane]; RHS[k][lane]=RHS[pk][lane]; RHS[pk][lane]=t; }
    }
    for (int k=0;k<32;k++)
      for (int i=k+1;i<32;i++)
        RHS[i][lane]=csub(RHS[i][lane], cmul(C1[i][k], RHS[k][lane]));
    for (int k=31;k>=0;k--){
      float2 xk=RHS[k][lane];
      for (int jj=k+1;jj<32;jj++) xk=csub(xk, cmul(C1[k][jj], RHS[jj][lane]));
      RHS[k][lane]=cdiv(xk, C1[k][k]);
    }
    if (col < 255)
      for (int i=0;i<32;i++) po[(size_t)b*8160 + (size_t)i*255 + col] = RHS[i][lane];
    if (blk==0 && lane==0)
      for (int i=0;i<32;i++) y0S[i]=RHS[i][0];
    __syncthreads();
  }

  // ---- y_adv[k][l] = Re( evecs[l][:] . (evals^k * y0) ) ----
  cS[lane]=y0S[lane];
  __syncthreads();
  for (int k=0;k<256;k++){
    float acc=0.f;
    for (int mm=0;mm<32;mm++){
      float2 e=C2[lane][mm], cc=cS[mm];
      acc += e.x*cc.x - e.y*cc.y;
    }
    yadv_out[(size_t)b*8192 + (size_t)k*32 + lane] = acc;
    __syncthreads();
    cS[lane]=cmul(cS[lane], evalS[lane]);
    __syncthreads();
  }

  // ---- store evals / evecs ----
  ((float2*)evals_out)[(size_t)b*32 + lane] = evalS[lane];
  float2* eo = (float2*)evecs_out;
  for (int i=0;i<32;i++) eo[(size_t)b*1024 + (size_t)i*32 + lane] = C2[i][lane];
}

// ================= launch =================
extern "C" void kernel_launch(void* const* d_in, const int* in_sizes, int n_in,
                              void* d_out, int out_size, void* d_ws, size_t ws_size,
                              hipStream_t stream)
{
  (void)in_sizes; (void)n_in; (void)out_size; (void)ws_size;
  const float* x       = (const float*)d_in[0];
  const float* eWin    = (const float*)d_in[1];
  const float* eBin    = (const float*)d_in[2];
  const float* eWh     = (const float*)d_in[3];
  const float* eBh     = (const float*)d_in[4];
  const float* eWout   = (const float*)d_in[5];
  const float* eBout   = (const float*)d_in[6];
  const float* dWin    = (const float*)d_in[7];
  const float* dBin    = (const float*)d_in[8];
  const float* dWh     = (const float*)d_in[9];
  const float* dBh     = (const float*)d_in[10];
  const float* dWout   = (const float*)d_in[11];
  const float* dBout   = (const float*)d_in[12];
  float* out = (float*)d_out;
  _Float16* ws = (_Float16*)d_ws;

  _Float16* pEncH = ws + WS_ENC_H;
  _Float16* pDecH = ws + WS_DEC_H;
  _Float16* pEncO = ws + WS_ENC_O;
  _Float16* pDecI = ws + WS_DEC_I;

  for (int l=0;l<4;l++){
    pack_w_f16<<<dim3(8,16), 32, 0, stream>>>(eWh + (size_t)l*65536, 256, 256, pEncH + (size_t)l*65536);
    pack_w_f16<<<dim3(8,16), 32, 0, stream>>>(dWh + (size_t)l*65536, 256, 256, pDecH + (size_t)l*65536);
  }
  pack_w_f16<<<dim3(8,2),  32, 0, stream>>>(eWout, 256, 32,  pEncO);
  pack_w_f16<<<dim3(1,16), 32, 0, stream>>>(dWin,  32,  256, pDecI);

  enc_kernel<<<2048, 128, MLP_LDS_BYTES, stream>>>(x, eWin, eBin, pEncH, eBh, pEncO, eBout, out + Y_OFF);
  dec_kernel<<<2048, 128, MLP_LDS_BYTES, stream>>>(out + Y_OFF, pDecI, dBin, pDecH, dBh, dWout, dBout, out + XAE_OFF);
  dmd_kernel<<<NBATCH, 32, 0, stream>>>(out + Y_OFF, out + EVALS_OFF, out + EVECS_OFF,
                                        out + PHI_OFF, out + YADV_OFF);
  dec_kernel<<<2048, 128, MLP_LDS_BYTES, stream>>>(out + YADV_OFF, pDecI, dBin, pDecH, dBh, dWout, dBout, out + XADV_OFF);
}
